// GraphSAGE_52218212385011
// MI455X (gfx1250) — matrix operand order repaired and checked by reading, not compiled, not run on hardware
//
#include <hip/hip_runtime.h>
#include <hip/hip_bf16.h>
#include <stdint.h>

// ---------------------------------------------------------------------------
// GraphSAGE 2-layer forward for MI455X (gfx1250), bf16 WMMA pipeline.
// ---------------------------------------------------------------------------

typedef __bf16 bf16_t;
typedef __attribute__((ext_vector_type(16))) __bf16 v16bf;
typedef __attribute__((ext_vector_type(8)))  __bf16 v8bf;
typedef __attribute__((ext_vector_type(4)))  __bf16 v4bf;
typedef __attribute__((ext_vector_type(8)))  float  v8f;
typedef __attribute__((ext_vector_type(4)))  float  v4f;

static constexpr int cU2   = 32768;
static constexpr int cN1   = 8192;
static constexpr int cN0   = 2048;
static constexpr int cFEAT = 512;
static constexpr int cE1   = 256;
static constexpr int cE2   = 128;

// LDS strides (bf16 elements); multiples of 8 so fragment loads are 16B-aligned.
static constexpr int SA  = 40;   // A tile stride (32 + 8 pad)
static constexpr int SB  = 40;   // B^T tile stride (32 + 8 pad)
static constexpr int ST1 = 520;  // T2 tile stride (512 + 8 pad)
static constexpr int ST2 = 264;  // T1 tile stride (256 + 8 pad)

union AB16 { v16bf v; v8bf h[2]; };

__device__ __forceinline__ unsigned wave_any(bool p) {
#if __has_builtin(__builtin_amdgcn_ballot_w32)
  return __builtin_amdgcn_ballot_w32(p);
#else
  return (unsigned)__ballot(p);
#endif
}

__device__ __forceinline__ v8f wmma_bf16(v16bf a, v16bf b, v8f c) {
  // D = A(16x32 bf16) * B(32x16 bf16) + C(16x16 f32)
  return __builtin_amdgcn_wmma_f32_16x16x32_bf16(
      /*neg_a=*/false, a, /*neg_b=*/false, b,
      /*c_mod=*/(short)0, c, /*reuse_a=*/false, /*reuse_b=*/false);
}

// A fragment (16x32 bf16) from row-major bf16 LDS tile.
// Lanes 0-15: row = mbase+lane, K = kbase+[0..15]
// Lanes 16-31: row = mbase+lane-16, K = kbase+[16..31]
__device__ __forceinline__ v16bf load_a_frag(const bf16_t* lds, int stride,
                                             int mbase, int kbase, int lane) {
  const int r  = mbase + (lane & 15);
  const int kb = kbase + ((lane >> 4) << 4);
  const bf16_t* p = lds + r * stride + kb;
  AB16 u;
  u.h[0] = *(const v8bf*)(p);
  u.h[1] = *(const v8bf*)(p + 8);
  return u.v;
}

// B fragment (32x16 bf16) from col-major bf16 LDS tile (lds[col][k]).
// Lanes 0-15: col = nbase+lane, K = [0..15]; lanes 16-31: col-16, K = [16..31]
__device__ __forceinline__ v16bf load_b_frag_lds(const bf16_t* lds, int stride,
                                                 int nbase, int lane) {
  const int c  = nbase + (lane & 15);
  const int kb = (lane >> 4) << 4;
  const bf16_t* p = lds + c * stride + kb;
  AB16 u;
  u.h[0] = *(const v8bf*)(p);
  u.h[1] = *(const v8bf*)(p + 8);
  return u.v;
}

// B fragment directly from global row-major fp32 weight W[n][ldw]:
// W row-major == W^T col-major, exactly the B layout. Convert fp32->bf16.
__device__ __forceinline__ v16bf load_b_frag_w(const float* __restrict__ w, int ldw,
                                               int nbase, int kk, int lane) {
  const int n  = nbase + (lane & 15);
  const int kb = kk + ((lane >> 4) << 4);
  const float* p = w + (size_t)n * ldw + kb;
  v4f f0 = *(const v4f*)(p);
  v4f f1 = *(const v4f*)(p + 4);
  v4f f2 = *(const v4f*)(p + 8);
  v4f f3 = *(const v4f*)(p + 12);
  v16bf b;
#pragma unroll
  for (int i = 0; i < 4; ++i) {
    b[i]      = (__bf16)f0[i];
    b[4 + i]  = (__bf16)f1[i];
    b[8 + i]  = (__bf16)f2[i];
    b[12 + i] = (__bf16)f3[i];
  }
  return b;
}

// B fragment directly from global row-major bf16 matrix g[n][ldg] (== B col-major).
__device__ __forceinline__ v16bf load_b_frag_g16(const bf16_t* __restrict__ g, int ldg,
                                                 int nbase, int kk, int lane) {
  const int n  = nbase + (lane & 15);
  const int kb = kk + ((lane >> 4) << 4);
  const bf16_t* p = g + (size_t)n * ldg + kb;
  AB16 u;
  u.h[0] = *(const v8bf*)(p);
  u.h[1] = *(const v8bf*)(p + 8);
  return u.v;
}

__device__ __forceinline__ v4bf cvt4(v4f f) {
  v4bf r;
  r[0] = (__bf16)f[0]; r[1] = (__bf16)f[1];
  r[2] = (__bf16)f[2]; r[3] = (__bf16)f[3];
  return r;
}

// ---------------------------------------------------------------------------
// Kernel 1: combined2^T = relu( (mask2 @ features) @ W2^T )^T   -> ws (bf16)
//   mask2 [N1,U2] fp32, features [U2,FEAT] fp32, W2 [E1,FEAT] fp32
//   output c2T [E1][N1] bf16 (transposed so kernel 2 reads it as B directly)
// Block: 64 rows of N1, 256 threads (8 waves in 2x4: 32 rows x 128 cols each).
// ---------------------------------------------------------------------------
struct alignas(16) SharedK1 {
  union {
    struct { bf16_t A[64 * SA]; bf16_t B[cFEAT * SB]; } s;
    bf16_t T[64 * ST1];
  } u;
  int wnz[8];
};

__global__ __launch_bounds__(256)
void sage_layer2_kernel(const float* __restrict__ features,
                        const float* __restrict__ weight2,
                        const float* __restrict__ mask2,
                        bf16_t* __restrict__ c2T) {
  __shared__ SharedK1 sh;
  const int tid  = threadIdx.x;
  const int lane = tid & 31;
  const int wid  = tid >> 5;
  const int mw   = wid >> 2;          // 0..1
  const int nw   = wid & 3;           // 0..3
  const int m0   = blockIdx.x * 64;

  // B-staging decomposition: 4x4 register-blocked transpose.
  const int bc4 = (tid & 127) * 4;    // column group (0..508)
  const int brh = (tid >> 7) * 16;    // row half (0 or 16)

  v8f acc[2][8];
#pragma unroll
  for (int i = 0; i < 2; ++i)
#pragma unroll
    for (int j = 0; j < 8; ++j) acc[i][j] = (v8f){};

  for (int kk = 0; kk < cU2; kk += 32) {
    // ---- stage A: mask2[m0..m0+63][kk..kk+31] -> bf16 LDS, detect all-zero tile
    bool nz = false;
#pragma unroll
    for (int i = 0; i < 2; ++i) {
      const int q  = tid * 2 + i;      // float4 index, 512 total
      const int r  = q >> 3;
      const int c4 = (q & 7) * 4;
      v4f f = *(const v4f*)(mask2 + (size_t)(m0 + r) * cU2 + kk + c4);
      nz |= (f[0] != 0.f) | (f[1] != 0.f) | (f[2] != 0.f) | (f[3] != 0.f);
      *(v4bf*)(&sh.u.s.A[r * SA + c4]) = cvt4(f);   // one ds_store_b64
    }
    if (lane == 0) sh.wnz[wid] = (wave_any(nz) != 0u);
    __syncthreads();
    const int any = sh.wnz[0] | sh.wnz[1] | sh.wnz[2] | sh.wnz[3] |
                    sh.wnz[4] | sh.wnz[5] | sh.wnz[6] | sh.wnz[7];
    if (any) {
      // ---- stage B: features[kk..kk+31][0..511] -> transposed bf16 LDS [col][k]
      // 4x4 register-block transpose: packed ds_store_b64, cvt_pk-friendly.
#pragma unroll
      for (int g = 0; g < 4; ++g) {
        const int r = brh + g * 4;
        const float* fp = features + (size_t)(kk + r) * cFEAT + bc4;
        v4f f0 = *(const v4f*)(fp);
        v4f f1 = *(const v4f*)(fp + cFEAT);
        v4f f2 = *(const v4f*)(fp + 2 * cFEAT);
        v4f f3 = *(const v4f*)(fp + 3 * cFEAT);
#pragma unroll
        for (int n = 0; n < 4; ++n) {
          v4bf pk;
          pk[0] = (__bf16)f0[n]; pk[1] = (__bf16)f1[n];
          pk[2] = (__bf16)f2[n]; pk[3] = (__bf16)f3[n];
          *(v4bf*)(&sh.u.s.B[(bc4 + n) * SB + r]) = pk;  // one ds_store_b64
        }
      }
      __syncthreads();
      v16bf a0 = load_a_frag(sh.u.s.A, SA, mw * 32 + 0,  0, lane);
      v16bf a1 = load_a_frag(sh.u.s.A, SA, mw * 32 + 16, 0, lane);
#pragma unroll
      for (int j = 0; j < 8; ++j) {
        v16bf b = load_b_frag_lds(sh.u.s.B, SB, nw * 128 + j * 16, lane);
        acc[0][j] = wmma_bf16(a0, b, acc[0][j]);
        acc[1][j] = wmma_bf16(a1, b, acc[1][j]);
      }
    }
    __syncthreads();
  }

  // ---- spill T2 tile (bf16) to LDS for the fused linear layer
#pragma unroll
  for (int mt = 0; mt < 2; ++mt) {
    const int mloc = mw * 32 + mt * 16 + ((lane >> 4) << 3);
#pragma unroll
    for (int j = 0; j < 8; ++j) {
      const int n = nw * 128 + j * 16 + (lane & 15);
#pragma unroll
      for (int r = 0; r < 8; ++r)
        sh.u.T[(mloc + r) * ST1 + n] = (__bf16)acc[mt][j][r];
    }
  }
  __syncthreads();

  // ---- GEMM2: relu(T2 @ W2^T), M=64, K=512, N=256 (nw covers 64 cols)
  v8f acc2[2][4];
#pragma unroll
  for (int i = 0; i < 2; ++i)
#pragma unroll
    for (int j = 0; j < 4; ++j) acc2[i][j] = (v8f){};

  for (int kk = 0; kk < cFEAT; kk += 32) {
    v16bf a0 = load_a_frag(sh.u.T, ST1, mw * 32 + 0,  kk, lane);
    v16bf a1 = load_a_frag(sh.u.T, ST1, mw * 32 + 16, kk, lane);
#pragma unroll
    for (int j = 0; j < 4; ++j) {
      v16bf b = load_b_frag_w(weight2, cFEAT, nw * 64 + j * 16, kk, lane);
      acc2[0][j] = wmma_bf16(a0, b, acc2[0][j]);
      acc2[1][j] = wmma_bf16(a1, b, acc2[1][j]);
    }
  }

  // ---- ReLU + store transposed: c2T[n][m], 8 contiguous rows per lane (B128)
#pragma unroll
  for (int mt = 0; mt < 2; ++mt) {
    const int m = m0 + mw * 32 + mt * 16 + ((lane >> 4) << 3);
#pragma unroll
    for (int j = 0; j < 4; ++j) {
      const int n = nw * 64 + j * 16 + (lane & 15);
      v8bf pk;
#pragma unroll
      for (int r = 0; r < 8; ++r)
        pk[r] = (__bf16)fmaxf(acc2[mt][j][r], 0.f);
      *(v8bf*)(c2T + (size_t)n * cN1 + m) = pk;
    }
  }
}

// ---------------------------------------------------------------------------
// Kernel 2: out = relu( (mask1 @ combined2) @ W1^T )    -> d_out (fp32)
//   mask1 [N0,N1] fp32, c2T [E1][N1] bf16 (B col-major: loaded directly), W1 [E2,E1]
// Block: 64 rows of N0, 256 threads (2x4 waves: 32 rows x 64 cols of E1).
// ---------------------------------------------------------------------------
struct alignas(16) SharedK2 {
  union {
    bf16_t A[64 * SA];
    bf16_t T[64 * ST2];
  } u;
  int wnz[8];
};

__global__ __launch_bounds__(256)
void sage_layer1_kernel(const float* __restrict__ mask1,
                        const bf16_t* __restrict__ c2T,
                        const float* __restrict__ weight1,
                        float* __restrict__ out) {
  __shared__ SharedK2 sh;
  const int tid  = threadIdx.x;
  const int lane = tid & 31;
  const int wid  = tid >> 5;
  const int mw   = wid >> 2;          // 0..1
  const int nw   = wid & 3;           // 0..3
  const int m0   = blockIdx.x * 64;

  v8f acc[2][4];
#pragma unroll
  for (int i = 0; i < 2; ++i)
#pragma unroll
    for (int j = 0; j < 4; ++j) acc[i][j] = (v8f){};

  for (int kk = 0; kk < cN1; kk += 32) {
    bool nz = false;
#pragma unroll
    for (int i = 0; i < 2; ++i) {
      const int q  = tid * 2 + i;
      const int r  = q >> 3;
      const int c4 = (q & 7) * 4;
      v4f f = *(const v4f*)(mask1 + (size_t)(m0 + r) * cN1 + kk + c4);
      nz |= (f[0] != 0.f) | (f[1] != 0.f) | (f[2] != 0.f) | (f[3] != 0.f);
      *(v4bf*)(&sh.u.A[r * SA + c4]) = cvt4(f);     // one ds_store_b64
    }
    if (lane == 0) sh.wnz[wid] = (wave_any(nz) != 0u);
    __syncthreads();
    const int any = sh.wnz[0] | sh.wnz[1] | sh.wnz[2] | sh.wnz[3] |
                    sh.wnz[4] | sh.wnz[5] | sh.wnz[6] | sh.wnz[7];
    if (any) {
      v16bf a0 = load_a_frag(sh.u.A, SA, mw * 32 + 0,  0, lane);
      v16bf a1 = load_a_frag(sh.u.A, SA, mw * 32 + 16, 0, lane);
#pragma unroll
      for (int j = 0; j < 4; ++j) {
        // B directly from global bf16 (c2T row n = column n of combined2)
        v16bf b = load_b_frag_g16(c2T, cN1, nw * 64 + j * 16, kk, lane);
        acc[0][j] = wmma_bf16(a0, b, acc[0][j]);
        acc[1][j] = wmma_bf16(a1, b, acc[1][j]);
      }
    }
    __syncthreads();
  }

  // ---- spill T1 tile (bf16) to LDS
#pragma unroll
  for (int mt = 0; mt < 2; ++mt) {
    const int mloc = mw * 32 + mt * 16 + ((lane >> 4) << 3);
#pragma unroll
    for (int j = 0; j < 4; ++j) {
      const int n = nw * 64 + j * 16 + (lane & 15);
#pragma unroll
      for (int r = 0; r < 8; ++r)
        sh.u.T[(mloc + r) * ST2 + n] = (__bf16)acc[mt][j][r];
    }
  }
  __syncthreads();

  // ---- GEMM2: relu(T1 @ W1^T), M=64, K=256, N=128 (nw covers 32 cols)
  v8f acc2[2][2];
#pragma unroll
  for (int i = 0; i < 2; ++i)
#pragma unroll
    for (int j = 0; j < 2; ++j) acc2[i][j] = (v8f){};

  for (int kk = 0; kk < cE1; kk += 32) {
    v16bf a0 = load_a_frag(sh.u.T, ST2, mw * 32 + 0,  kk, lane);
    v16bf a1 = load_a_frag(sh.u.T, ST2, mw * 32 + 16, kk, lane);
#pragma unroll
    for (int j = 0; j < 2; ++j) {
      v16bf b = load_b_frag_w(weight1, cE1, nw * 32 + j * 16, kk, lane);
      acc2[0][j] = wmma_bf16(a0, b, acc2[0][j]);
      acc2[1][j] = wmma_bf16(a1, b, acc2[1][j]);
    }
  }

  // ---- ReLU + store fp32 out[m][n]
#pragma unroll
  for (int mt = 0; mt < 2; ++mt) {
    const int mg = m0 + mw * 32 + mt * 16 + ((lane >> 4) << 3);
#pragma unroll
    for (int j = 0; j < 2; ++j) {
      const int n = nw * 32 + j * 16 + (lane & 15);
#pragma unroll
      for (int r = 0; r < 8; ++r)
        out[(size_t)(mg + r) * cE2 + n] = fmaxf(acc2[mt][j][r], 0.f);
    }
  }
}

// ---------------------------------------------------------------------------
extern "C" void kernel_launch(void* const* d_in, const int* in_sizes, int n_in,
                              void* d_out, int out_size, void* d_ws, size_t ws_size,
                              hipStream_t stream) {
  (void)in_sizes; (void)n_in; (void)out_size; (void)ws_size;
  const float* features = (const float*)d_in[0];   // [U2, FEAT]
  const float* weight2  = (const float*)d_in[1];   // [E1, FEAT]
  const float* weight1  = (const float*)d_in[2];   // [E2, E1]
  const float* mask2    = (const float*)d_in[3];   // [N1, U2]
  const float* mask1    = (const float*)d_in[4];   // [N0, N1]
  float*  out = (float*)d_out;                     // [N0, E2]
  bf16_t* c2T = (bf16_t*)d_ws;                     // [E1][N1] bf16 = 4 MB scratch

  sage_layer2_kernel<<<cN1 / 64, 256, 0, stream>>>(features, weight2, mask2, c2T);
  sage_layer1_kernel<<<cN0 / 64, 256, 0, stream>>>(mask1, c2T, weight1, out);
}
